// ConvNeXtBlock_90649579749501
// MI455X (gfx1250) — compile-verified, matrix operand order checked by reading, not run on hardware
//
#include <hip/hip_runtime.h>
#include <hip/hip_bf16.h>
#include <math.h>

#define DIMC   128
#define HID    512
#define HW     56
#define NBATCH 32
#define TOKENS (NBATCH*HW*HW)   // 100352
#define EPSLN  1e-6f

typedef __attribute__((ext_vector_type(16))) __bf16 bf16x16;
typedef __attribute__((ext_vector_type(8)))  float  f32x8;
typedef __attribute__((ext_vector_type(4)))  unsigned int v4u;
typedef __attribute__((ext_vector_type(8)))  int  v8i;
typedef __attribute__((ext_vector_type(4)))  int  v4i;

#if __has_builtin(__builtin_amdgcn_tensor_load_to_lds)
#define HAVE_TDM 1
#else
#define HAVE_TDM 0
#endif

#if __has_builtin(__builtin_amdgcn_sched_barrier)
#define SCHED_FENCE() __builtin_amdgcn_sched_barrier(0)
#else
#define SCHED_FENCE()
#endif

static __device__ __forceinline__ unsigned short f32_to_bf16_rne(float f) {
    union { float f; unsigned u; } v; v.f = f;
    unsigned r = v.u + 0x7FFFu + ((v.u >> 16) & 1u);
    return (unsigned short)(r >> 16);
}

// low 32 bits of a generic pointer to LDS == wave-relative LDS byte address
static __device__ __forceinline__ unsigned lds_off(const void* p) {
    return (unsigned)(unsigned long long)(uintptr_t)p;
}

#if HAVE_TDM
// Tensor Data Mover: 2D tile load, D# per ISA 8.3/8.4.
// dims/strides in data_size units; data_size_log2: 0=1B,1=2B,2=4B.
static __device__ __forceinline__ void
tdm_load_2d(unsigned lds_addr, const void* gptr,
            unsigned tensor_d0, unsigned tensor_d1,
            unsigned tile_d0, unsigned tile_d1,
            unsigned long long d0_stride, unsigned data_size_log2)
{
    unsigned long long ga = (unsigned long long)(uintptr_t)gptr;
    v4u g0;
    g0.x = 1u;                                     // count=1, user desc
    g0.y = lds_addr;                               // lds_addr -> bits[63:32]
    g0.z = (unsigned)(ga & 0xFFFFFFFFu);           // global_addr[31:0]
    g0.w = (unsigned)((ga >> 32) & 0x01FFFFFFu)    // global_addr[56:32]
         | (2u << 30);                             // type=2 ("image")
    v8i g1;
    g1[0] = (int)(data_size_log2 << 16);                           // wg_mask=0 | data_size
    g1[1] = (int)((tensor_d0 & 0xFFFFu) << 16);                    // bar_addr=0 | dim0.lo
    g1[2] = (int)((tensor_d0 >> 16) | ((tensor_d1 & 0xFFFFu) << 16));
    g1[3] = (int)((tensor_d1 >> 16) | (tile_d0 << 16));            // dim1.hi | tile_dim0
    g1[4] = (int)(tile_d1 & 0xFFFFu);                              // tile_dim1 | tile_dim2=0
    g1[5] = (int)(d0_stride & 0xFFFFFFFFull);                      // dim0_stride lo
    g1[6] = (int)((d0_stride >> 32) & 0xFFFFull);                  // dim0_stride hi
    g1[7] = 0;
    v4i z4 = {0,0,0,0};
#if __clang_major__ >= 23
    v8i z8 = {0,0,0,0,0,0,0,0};
    __builtin_amdgcn_tensor_load_to_lds(g0, g1, z4, z4, z8, 0);
#else
    __builtin_amdgcn_tensor_load_to_lds(g0, g1, z4, z4, 0);
#endif
}
#endif

// packed-dword fragment load matching 16-bit WMMA A/B VGPR layout (ISA 7.12.2)
static __device__ __forceinline__ void
load_frag(unsigned int* u, const unsigned short* rowbase, int kbase)
{
#pragma unroll
    for (int j = 0; j < 8; ++j) {
        const int k = kbase + (j < 4 ? 2*j : 2*j + 8);
        u[j] = *(const unsigned int*)&rowbase[k];
    }
}

union Frag { bf16x16 v; unsigned int u[8]; };

// ---------------------------------------------------------------------------
// Kernel 0: one-shot weight prep.  w1[128][512] -> W1T bf16 [512][128]
//           w2[512][128] -> W2C bf16 [16 chunks][128 c][32 f-local]
// Outputs are the exact LDS-resident layouts of kernel 2, so every per-chunk
// weight load becomes a contiguous 8KB DMA.
// ---------------------------------------------------------------------------
__global__ void __launch_bounds__(256)
weight_prep_kernel(const float* __restrict__ w1, const float* __restrict__ w2,
                   unsigned short* __restrict__ W1T,
                   unsigned short* __restrict__ W2C)
{
    const int i = blockIdx.x * 256 + threadIdx.x;   // 0..65535
    {
        const int f = i >> 7, c = i & 127;
        W1T[i] = f32_to_bf16_rne(w1[c*HID + f]);
    }
    {
        const int fc = i >> 12, r = i & 4095;
        const int c = r >> 5, fl = r & 31;
        W2C[i] = f32_to_bf16_rne(w2[(fc*32 + fl)*DIMC + c]);
    }
}

// ---------------------------------------------------------------------------
// Kernel 1: depthwise 7x7 conv + bias + LayerNorm(channels) -> bf16 A[T,128]
// One wave32 per token; 4 channels per lane; wave-shuffle LN reduction.
// x (51MB) is L2-resident, so the 49x conv re-reads stay on chip.
// ---------------------------------------------------------------------------
__global__ void __launch_bounds__(256)
dwconv_ln_kernel(const float* __restrict__ x,
                 const float* __restrict__ dw_w,
                 const float* __restrict__ dw_b,
                 const float* __restrict__ ln_g,
                 const float* __restrict__ ln_b,
                 unsigned short* __restrict__ A)
{
    const int wave = threadIdx.x >> 5;
    const int lane = threadIdx.x & 31;
    const int tok  = blockIdx.x * 8 + wave;
    const int n  = tok / (HW*HW);
    const int hw = tok - n*(HW*HW);
    const int h  = hw / HW;
    const int w  = hw - h*HW;

    float v[4];
    float s1 = 0.f, s2 = 0.f;
#pragma unroll
    for (int cc = 0; cc < 4; ++cc) {
        const int c = lane*4 + cc;
        const float* xc = x + ((size_t)n*DIMC + c) * (HW*HW);
        const float* wc = dw_w + c*49;
        float acc = dw_b[c];
#pragma unroll
        for (int i = 0; i < 7; ++i) {
            const int hh = h + i - 3;
            if (hh < 0 || hh >= HW) continue;
#pragma unroll
            for (int j = 0; j < 7; ++j) {
                const int ww = w + j - 3;
                if (ww < 0 || ww >= HW) continue;
                acc = fmaf(xc[hh*HW + ww], wc[i*7 + j], acc);
            }
        }
        v[cc] = acc;
        s1 += acc;
        s2 = fmaf(acc, acc, s2);
    }
#pragma unroll
    for (int off = 16; off > 0; off >>= 1) {
        s1 += __shfl_xor(s1, off, 32);
        s2 += __shfl_xor(s2, off, 32);
    }
    const float mean = s1 * (1.0f/DIMC);
    const float var  = s2 * (1.0f/DIMC) - mean*mean;
    const float rstd = rsqrtf(var + EPSLN);

    unsigned short* Arow = A + (size_t)tok * DIMC;
#pragma unroll
    for (int cc = 0; cc < 4; ++cc) {
        const int c = lane*4 + cc;
        const float y = (v[cc] - mean) * rstd * ln_g[c] + ln_b[c];
        Arow[c] = f32_to_bf16_rne(y);
    }
}

// ---------------------------------------------------------------------------
// Kernel 2: fused MLP  (A @ W1 + b1) -> GELU -> (@ W2 + b2)*gamma + residual
// 128-token tile per block, 8 waves; hidden dim streamed in chunks of 32.
// Weight chunks are double-buffered and prefetched by the Tensor Data Mover
// one chunk ahead; s_wait_tensorcnt(2) exploits in-order TDM completion so
// the DMA of chunk fc+1 overlaps compute of chunk fc.
// Per GEMM stage: all fragments preloaded, a sched_barrier pins them above
// the matrix ops -> one DS wait, then an unbroken 8-deep WMMA burst.
// ---------------------------------------------------------------------------
__global__ void __launch_bounds__(256)
mlp_wmma_kernel(const unsigned short* __restrict__ A,    // [T][128] bf16
                const unsigned short* __restrict__ W1T,  // [512][128] bf16
                const unsigned short* __restrict__ W2C,  // [16][128][32] bf16
                const float* __restrict__ b1,
                const float* __restrict__ b2,
                const float* __restrict__ gamma,
                const float* __restrict__ x,             // residual, NCHW
                float* __restrict__ out)
{
    __shared__ unsigned short As [128][128];     // token x K(c)        32 KB
    __shared__ unsigned short B1s[2][32][128];   // N(f) x K(c), 2-buf  16 KB
    __shared__ unsigned short B2s[2][128][32];   // N(c) x K(f), 2-buf  16 KB
    __shared__ unsigned short Hs [128][32];      // token x K(f)         8 KB

    const int tid   = threadIdx.x;
    const int wave  = tid >> 5;
    const int lane  = tid & 31;
    const int lan16 = lane & 15;
    const int hi    = lane >> 4;
    const int tok0  = blockIdx.x * 128;

#if HAVE_TDM
    if (wave == 0) {    // A tile (32KB) + first weight chunk
        tdm_load_2d(lds_off(&As[0][0]), A + (size_t)tok0 * DIMC,
                    DIMC, TOKENS, DIMC, 128, DIMC, 1);
        tdm_load_2d(lds_off(&B1s[0][0][0]), W1T, DIMC, HID, DIMC, 32, DIMC, 1);
        tdm_load_2d(lds_off(&B2s[0][0][0]), W2C, 32, 16*DIMC, 32, DIMC, 32, 1);
    }
#else
    {
        const unsigned int* Ag = (const unsigned int*)(A + (size_t)tok0 * DIMC);
        unsigned int* Al = (unsigned int*)&As[0][0];
#pragma unroll
        for (int i = 0; i < 32; ++i)
            Al[tid + i*256] = Ag[tid + i*256];
    }
#endif

    f32x8 acc[8];
#pragma unroll
    for (int nb = 0; nb < 8; ++nb) acc[nb] = (f32x8){0,0,0,0,0,0,0,0};

    const int arow = wave*16 + lan16;

    for (int fc = 0; fc < 16; ++fc) {
        const int buf = fc & 1;
        __syncthreads();   // everyone done with the buffer we are about to fill
#if HAVE_TDM
        if (wave == 0) {
            if (fc < 15) {     // prefetch next chunk into the other buffer
                tdm_load_2d(lds_off(&B1s[buf ^ 1][0][0]),
                            W1T + (size_t)(fc + 1)*32*DIMC,
                            DIMC, HID, DIMC, 32, DIMC, 1);
                tdm_load_2d(lds_off(&B2s[buf ^ 1][0][0]),
                            W2C + (size_t)(fc + 1)*DIMC*32,
                            32, 16*DIMC, 32, DIMC, 32, 1);
                // in-order TDM: <=2 outstanding => everything for fc landed
                __builtin_amdgcn_s_wait_tensorcnt(2);
            } else {
                __builtin_amdgcn_s_wait_tensorcnt(0);
            }
        }
#else
        {
            const unsigned int* g1p = (const unsigned int*)(W1T + (size_t)fc*32*DIMC);
            const unsigned int* g2p = (const unsigned int*)(W2C + (size_t)fc*DIMC*32);
            unsigned int* l1 = (unsigned int*)&B1s[buf][0][0];
            unsigned int* l2 = (unsigned int*)&B2s[buf][0][0];
#pragma unroll
            for (int i = 0; i < 8; ++i) {
                l1[tid + i*256] = g1p[tid + i*256];
                l2[tid + i*256] = g2p[tid + i*256];
            }
        }
#endif
        __syncthreads();   // buf ready

        // ---- GEMM1: H[16 tok x 32 f] = A[16x128] @ W1chunk ----
        Frag af[4], bf1[8];
#pragma unroll
        for (int ks = 0; ks < 4; ++ks)
            load_frag(af[ks].u, &As[arow][0], ks*32 + hi*8);
#pragma unroll
        for (int p = 0; p < 8; ++p) {
            const int ks = p >> 1, nb2 = p & 1;
            load_frag(bf1[p].u, &B1s[buf][nb2*16 + lan16][0], ks*32 + hi*8);
        }
        SCHED_FENCE();   // pin all ds_loads above the WMMA burst
        f32x8 hacc[2];
        hacc[0] = (f32x8){0,0,0,0,0,0,0,0};
        hacc[1] = (f32x8){0,0,0,0,0,0,0,0};
#pragma unroll
        for (int p = 0; p < 8; ++p) {
            const int ks = p >> 1, nb2 = p & 1;
            hacc[nb2] = __builtin_amdgcn_wmma_f32_16x16x32_bf16(
                false, af[ks].v, false, bf1[p].v, (short)0, hacc[nb2],
                false, false);
        }
        SCHED_FENCE();

        // bias + exact GELU, write bf16 chunk to Hs (A-layout for GEMM2)
#pragma unroll
        for (int nb2 = 0; nb2 < 2; ++nb2) {
            const float bias = b1[fc*32 + nb2*16 + lan16];
#pragma unroll
            for (int vv = 0; vv < 8; ++vv) {
                float hv = hacc[nb2][vv] + bias;
                hv = 0.5f * hv * (1.0f + erff(hv * 0.70710678118654752f));
                Hs[wave*16 + vv + hi*8][nb2*16 + lan16] = f32_to_bf16_rne(hv);
            }
        }
        __syncthreads();

        // ---- GEMM2: acc[16 tok x 128 c] += H[16x32] @ W2chunk ----
        Frag ha, bf2[8];
        load_frag(ha.u, &Hs[arow][0], hi*8);
#pragma unroll
        for (int nb = 0; nb < 8; ++nb)
            load_frag(bf2[nb].u, &B2s[buf][nb*16 + lan16][0], hi*8);
        SCHED_FENCE();   // pin all ds_loads above the WMMA burst
#pragma unroll
        for (int nb = 0; nb < 8; ++nb)
            acc[nb] = __builtin_amdgcn_wmma_f32_16x16x32_bf16(
                false, ha.v, false, bf2[nb].v, (short)0, acc[nb],
                false, false);
        SCHED_FENCE();
    }

    // ---- epilogue: out = x + gamma*(acc + b2), scattered to NCHW ----
#pragma unroll
    for (int nb = 0; nb < 8; ++nb) {
        const int c  = nb*16 + lan16;
        const float g  = gamma[c];
        const float bb = b2[c];
#pragma unroll
        for (int vv = 0; vv < 8; ++vv) {
            const int t  = tok0 + wave*16 + vv + hi*8;
            const int n  = t / (HW*HW);
            const int hw = t - n*(HW*HW);
            const size_t idx = ((size_t)n*DIMC + c)*(HW*HW) + hw;
            out[idx] = x[idx] + g * (acc[nb][vv] + bb);
        }
    }
}

// ---------------------------------------------------------------------------
extern "C" void kernel_launch(void* const* d_in, const int* in_sizes, int n_in,
                              void* d_out, int out_size, void* d_ws, size_t ws_size,
                              hipStream_t stream) {
    const float* x     = (const float*)d_in[0];
    const float* dw_w  = (const float*)d_in[1];
    const float* dw_b  = (const float*)d_in[2];
    const float* ln_g  = (const float*)d_in[3];
    const float* ln_b  = (const float*)d_in[4];
    const float* w1    = (const float*)d_in[5];
    const float* b1    = (const float*)d_in[6];
    const float* w2    = (const float*)d_in[7];
    const float* b2    = (const float*)d_in[8];
    const float* gamma = (const float*)d_in[9];
    float* out = (float*)d_out;

    // workspace: A [T][128] bf16 (25.7MB) | W1T 128KB | W2C 128KB
    unsigned short* A   = (unsigned short*)d_ws;
    unsigned short* W1T = A + (size_t)TOKENS * DIMC;
    unsigned short* W2C = W1T + (size_t)HID * DIMC;

    weight_prep_kernel<<<256, 256, 0, stream>>>(w1, w2, W1T, W2C);
    dwconv_ln_kernel<<<TOKENS/8, 256, 0, stream>>>(x, dw_w, dw_b, ln_g, ln_b, A);
    mlp_wmma_kernel<<<TOKENS/128, 256, 0, stream>>>(A, W1T, W2C, b1, b2, gamma, x, out);
}
